// Int4WeightInt8ActConv2d_7980049236643
// MI455X (gfx1250) — 8-bit matrix operand order repaired and checked by reading, compiled, not run on hardware
//
#include <hip/hip_runtime.h>


typedef __attribute__((ext_vector_type(8))) int   v8i;
typedef __attribute__((ext_vector_type(4))) int   v4i;
typedef __attribute__((ext_vector_type(2))) int   v2i;
typedef __attribute__((ext_vector_type(4))) float v4f;

#define N_IMG   32
#define C_IN    256
#define H_IN    56
#define W_IN    56
#define C_OUT   256
#define H_OUT   28
#define W_OUT   28
#define OHW     (H_OUT * W_OUT)
#define GEMM_M  (N_IMG * OHW)
#define GEMM_N  C_OUT
#define GEMM_K  (C_IN * 9)

#define BM      128
#define BN      64
#define BK      64
#define KSTEPS  (GEMM_K / BK)
#define LDSROW  80

#define A_INV_SCALE 20.0f
#define OUT_SCALE   0.0005f

#if __has_builtin(__builtin_amdgcn_global_load_async_to_lds_b128)
#define HAS_ASYNC_LDS 1
#else
#define HAS_ASYNC_LDS 0
#endif

__device__ __forceinline__ void async_copy16(const unsigned char* g, unsigned char* l)
{
#if HAS_ASYNC_LDS
    __builtin_amdgcn_global_load_async_to_lds_b128(
        (__attribute__((address_space(1))) v4i*)(unsigned char*)g,
        (__attribute__((address_space(3))) v4i*)l,
        0, 0);
#else
    *(v4i*)l = *(const v4i*)g;
#endif
}

__device__ __forceinline__ void wait_async_all()
{
#if HAS_ASYNC_LDS
#if __has_builtin(__builtin_amdgcn_s_wait_asynccnt)
    __builtin_amdgcn_s_wait_asynccnt(0);
#else
    asm volatile("s_wait_asynccnt 0x0" ::: "memory");
#endif
#endif
}

__global__ __launch_bounds__(256) void quant_im2col_kernel(
    const float* __restrict__ x, unsigned char* __restrict__ aq)
{
    const int tid = blockIdx.x * 256 + threadIdx.x;
    const int m   = tid / (GEMM_K / 4);
    const int k4  = (tid - m * (GEMM_K / 4)) * 4;

    const int n  = m / OHW;
    const int r  = m - n * OHW;
    const int oh = r / W_OUT;
    const int ow = r - oh * W_OUT;

    unsigned int packed = 0;
#pragma unroll
    for (int i = 0; i < 4; ++i) {
        const int k  = k4 + i;
        const int ci = k / 9;
        const int t  = k - ci * 9;
        const int kh = t / 3;
        const int kw = t - kh * 3;
        const int ih = oh * 2 - 1 + kh;
        const int iw = ow * 2 - 1 + kw;
        int q = 0;
        if ((unsigned)ih < (unsigned)H_IN && (unsigned)iw < (unsigned)W_IN) {
            const float v = x[(((long)n * C_IN + ci) * H_IN + ih) * W_IN + iw];
            float s = __builtin_rintf(v * A_INV_SCALE);
            s = fminf(fmaxf(s, -128.0f), 127.0f);
            q = (int)s;
        }
        packed |= ((unsigned int)(q & 0xFF)) << (8 * i);
    }
    ((unsigned int*)aq)[tid] = packed;
}

__global__ __launch_bounds__(256) void pack_weights_kernel(
    const float* __restrict__ w, unsigned char* __restrict__ bq)
{
    const int tid = blockIdx.x * 256 + threadIdx.x;
    unsigned int packed = 0;
#pragma unroll
    for (int i = 0; i < 4; ++i) {
        const int q = (int)w[(long)tid * 4 + i];
        packed |= ((unsigned int)(q & 0xFF)) << (8 * i);
    }
    ((unsigned int*)bq)[tid] = packed;
}

__device__ __forceinline__ v8i load_a8_frag(const unsigned char* p)
{
    v4i d0 = *(const v4i*)(p);
    v4i d1 = *(const v4i*)(p + 32);
    v8i r = { d0.x, d0.y, d0.z, d0.w, d1.x, d1.y, d1.z, d1.w };
    return r;
}

__device__ __forceinline__ v8i load_b8_frag(const unsigned char* p)
{
    v4i d0 = *(const v4i*)(p);
    v4i d1 = *(const v4i*)(p + 32);
    v8i r = { d0.x, d0.y, d0.z, d0.w, d1.x, d1.y, d1.z, d1.w };
    return r;
}

__device__ __forceinline__ void store_tile(const v8i& acc, int m0, int ncol,
                                           float* __restrict__ out)
{
    const int img = m0 / OHW;
    const int rem = m0 - img * OHW;
    float* p = out + ((long)img * C_OUT + ncol) * OHW + rem;
    v4f lo = { (float)acc[0] * OUT_SCALE, (float)acc[1] * OUT_SCALE,
               (float)acc[2] * OUT_SCALE, (float)acc[3] * OUT_SCALE };
    v4f hi = { (float)acc[4] * OUT_SCALE, (float)acc[5] * OUT_SCALE,
               (float)acc[6] * OUT_SCALE, (float)acc[7] * OUT_SCALE };
    *(v4f*)(p)     = lo;
    *(v4f*)(p + 4) = hi;
}

__global__ __launch_bounds__(256) void conv_wmma_iu8_kernel(
    const unsigned char* __restrict__ aq,
    const unsigned char* __restrict__ bq,
    float* __restrict__ out)
{
    __shared__ __align__(16) unsigned char As[2][BM * LDSROW];
    __shared__ __align__(16) unsigned char Bs[2][BN * LDSROW];

    const int t    = threadIdx.x;
    const int lane = t & 31;
    const int hi   = lane >> 4;
    const int l15  = lane & 15;
    const int wave = t >> 5;
    const int wm   = wave & 3;
    const int wn   = wave >> 2;

    const int blockM = blockIdx.x * BM;
    const int blockN = blockIdx.y * BN;

    const int sr = t >> 2;
    const int so = (t & 3) * 16;
    const unsigned char* gA = aq + (long)(blockM + sr) * GEMM_K + so;
    const unsigned char* gB = bq + (long)(blockN + sr) * GEMM_K + so;
    const int sAoff = sr * LDSROW + so;
    const int sBoff = sr * LDSROW + so;

    const int fAoff = (wm * 32 + l15) * LDSROW + hi * 16;
    const int fBoff = (wn * 32 + l15) * LDSROW + hi * 16;

    v8i acc00 = {0,0,0,0,0,0,0,0};
    v8i acc01 = {0,0,0,0,0,0,0,0};
    v8i acc10 = {0,0,0,0,0,0,0,0};
    v8i acc11 = {0,0,0,0,0,0,0,0};

    {
        async_copy16(gA,                     &As[0][sAoff]);
        async_copy16(gA + (long)64 * GEMM_K, &As[0][sAoff + 64 * LDSROW]);
        async_copy16(gB,                     &Bs[0][sBoff]);
    }
    wait_async_all();
    __syncthreads();

    for (int ks = 0; ks < KSTEPS; ++ks) {
        const int cur = ks & 1;

        if (ks + 1 < KSTEPS) {
            const unsigned char* ga = gA + (ks + 1) * BK;
            const unsigned char* gb = gB + (ks + 1) * BK;
            async_copy16(ga,                     &As[cur ^ 1][sAoff]);
            async_copy16(ga + (long)64 * GEMM_K, &As[cur ^ 1][sAoff + 64 * LDSROW]);
            async_copy16(gb,                     &Bs[cur ^ 1][sBoff]);
        }

        const unsigned char* fA = &As[cur][fAoff];
        const unsigned char* fB = &Bs[cur][fBoff];
        const v8i a0 = load_a8_frag(fA);
        const v8i a1 = load_a8_frag(fA + 16 * LDSROW);
        const v8i b0 = load_b8_frag(fB);
        const v8i b1 = load_b8_frag(fB + 16 * LDSROW);

        acc00 = __builtin_amdgcn_wmma_i32_16x16x64_iu8(true, a0, true, b0, acc00, false, false);
        acc01 = __builtin_amdgcn_wmma_i32_16x16x64_iu8(true, a0, true, b1, acc01, false, false);
        acc10 = __builtin_amdgcn_wmma_i32_16x16x64_iu8(true, a1, true, b0, acc10, false, false);
        acc11 = __builtin_amdgcn_wmma_i32_16x16x64_iu8(true, a1, true, b1, acc11, false, false);

        if (ks + 1 < KSTEPS) {
            wait_async_all();
            __syncthreads();
        }
    }

    const int mbase = blockM + wm * 32 + hi * 8;
    const int nbase = blockN + wn * 32 + l15;
    store_tile(acc00, mbase,      nbase,      out);
    store_tile(acc01, mbase,      nbase + 16, out);
    store_tile(acc10, mbase + 16, nbase,      out);
    store_tile(acc11, mbase + 16, nbase + 16, out);
}

extern "C" void kernel_launch(void* const* d_in, const int* in_sizes, int n_in,
                              void* d_out, int out_size, void* d_ws, size_t ws_size,
                              hipStream_t stream)
{
    (void)in_sizes; (void)n_in; (void)out_size; (void)ws_size;

    const float* x = (const float*)d_in[0];
    const float* w = (const float*)d_in[1];
    float* out     = (float*)d_out;

    unsigned char* aq = (unsigned char*)d_ws;
    unsigned char* bq = aq + (size_t)GEMM_M * GEMM_K;

    {
        const int total = GEMM_M * (GEMM_K / 4);
        quant_im2col_kernel<<<total / 256, 256, 0, stream>>>(x, aq);
    }
    {
        const int total = (GEMM_N * GEMM_K) / 4;
        pack_weights_kernel<<<total / 256, 256, 0, stream>>>(w, bq);
    }
    {
        dim3 grid(GEMM_M / BM, GEMM_N / BN);
        conv_wmma_iu8_kernel<<<grid, 256, 0, stream>>>(aq, bq, out);
    }
}
